// DIN_39616778338785
// MI455X (gfx1250) — matrix operand order repaired and checked by reading, not compiled, not run on hardware
//
#include <hip/hip_runtime.h>

// ---------------- types ----------------
typedef __attribute__((ext_vector_type(16))) __bf16 v16bf;
typedef __attribute__((ext_vector_type(8)))  __bf16 v8bf;
typedef __attribute__((ext_vector_type(8)))  float  v8f;

#define BB   32
#define TT   200
#define TTP  208      // T padded to 13*16
#define TQ   50
#define EE   64
#define UU   36
#define UUP  48       // attention hidden padded to 3*16
#define ROWS (BB*TQ)  // 1600 trunk rows

// ---------------- prep: weight decomposition + BN folding + fragment swizzle ----------------
// Fragment order for a KxN bf16 B-operand: index = ((kt*NT + nt)*32 + lane)*16 + j
// where element = W[kt*32 + 8*(lane/16) + (j&7) + 16*(j>>3)][nt*16 + lane%16]  (ISA 7.12.2 B layout)
struct PrepArgs {
  const float *W1, *W2, *dice_a, *dice_m, *dice_v;
  const float *bn1_g,*bn1_b,*bn1_m,*bn1_v,*Wm1,*bm1;
  const float *bn2_g,*bn2_b,*bn2_m,*bn2_v,*Wm2,*bm2;
  const float *bn3_g,*bn3_b,*bn3_m,*bn3_v,*Wm3,*bm3;
  float *Wq,*Wk,*W2p,*da,*dm,*dv,*b1p,*b2p,*b3p;
  __bf16 *WpF,*W1p,*W2mp,*W3p;
};

__device__ void fold_bn_swz(int tid, const float* g, const float* bb, const float* m,
                            const float* v, const float* W, const float* bias,
                            int K, int N, __bf16* Wout, float* bout) {
  int ntiles = N >> 4;
  for (int i = tid; i < K*N; i += 256) {
    int j    = i & 15;
    int lane = (i >> 4) & 31;
    int tile = i >> 9;                       // kt*ntiles + nt
    int kt = tile / ntiles, nt = tile % ntiles;
    int f = kt*32 + 8*(lane >> 4) + (j & 7) + 16*(j >> 3);
    int o = nt*16 + (lane & 15);
    float s = g[f] * rsqrtf(v[f] + 1e-6f);
    Wout[i] = (__bf16)(s * W[f*N + o]);
  }
  for (int o = tid; o < N; o += 256) {
    float acc = bias[o];
    for (int f = 0; f < K; ++f) {
      float s = g[f] * rsqrtf(v[f] + 1e-6f);
      float t = bb[f] - m[f] * s;
      acc += t * W[f*N + o];
    }
    bout[o] = acc;
  }
}

__global__ __launch_bounds__(256) void prep_kernel(PrepArgs A) {
  int tid = threadIdx.x;
  // W1 split: rows [0:64)=Wa(q), [64:128)=Wb(k), [128:192)=Wc(q-k), [192:256)=Wd(q*k)
  for (int i = tid; i < EE*UUP; i += 256) {
    int k = i / UUP, u = i % UUP;
    float wq = 0.f, wk = 0.f;
    if (u < UU) {
      float wa = A.W1[k*UU + u];
      float wb = A.W1[(64 + k)*UU + u];
      float wc = A.W1[(128 + k)*UU + u];
      wq = wa + wc;
      wk = wb - wc;
    }
    A.Wq[i] = wq;
    A.Wk[i] = wk;
  }
  // W1d (q*k block) in WMMA B-fragment order: [nt][ks][lane][j], 3*2*32*16 elems
  for (int i = tid; i < 3*2*32*16; i += 256) {
    int j = i & 15, lane = (i >> 4) & 31, ks = (i >> 9) & 1, nt = i >> 10;
    int k = ks*32 + 8*(lane >> 4) + (j & 7) + 16*(j >> 3);
    int n = nt*16 + (lane & 15);
    float wp = (n < UU) ? A.W1[(192 + k)*UU + n] : 0.f;
    A.WpF[i] = (__bf16)wp;
  }
  if (tid < UUP) {
    bool ok = tid < UU;
    A.W2p[tid] = ok ? A.W2[tid]     : 0.f;
    A.da[tid]  = ok ? A.dice_a[tid] : 0.f;
    A.dm[tid]  = ok ? A.dice_m[tid] : 0.f;
    A.dv[tid]  = ok ? A.dice_v[tid] : 1.f;
  }
  fold_bn_swz(tid, A.bn1_g, A.bn1_b, A.bn1_m, A.bn1_v, A.Wm1, A.bm1, 256, 256, A.W1p,  A.b1p);
  fold_bn_swz(tid, A.bn2_g, A.bn2_b, A.bn2_m, A.bn2_v, A.Wm2, A.bm2, 256, 128, A.W2mp, A.b2p);
  fold_bn_swz(tid, A.bn3_g, A.bn3_b, A.bn3_m, A.bn3_v, A.Wm3, A.bm3, 128,  64, A.W3p,  A.b3p);
}

// ---------------- Q'/K' precompute: ub@Wq (+b1), items@Wk ----------------
__global__ __launch_bounds__(256) void qk_kernel(const float* __restrict__ ub,
                                                 const float* __restrict__ items,
                                                 const float* __restrict__ Wq,
                                                 const float* __restrict__ Wk,
                                                 const float* __restrict__ b1,
                                                 float* __restrict__ Qp,
                                                 float* __restrict__ Kp) {
  int idx = blockIdx.x * 256 + threadIdx.x;
  const int NQ = BB*TTP*UUP;
  const int NK = BB*TQ*UUP;
  if (idx < NQ) {
    int u = idx % UUP, row = idx / UUP;
    int t = row % TTP, b = row / TTP;
    float val = 0.f;
    if (t < TT && u < UU) {
      const float* x = ub + (b*TT + t)*EE;
      for (int k = 0; k < EE; ++k) val += x[k] * Wq[k*UUP + u];
      val += b1[u];
    }
    Qp[idx] = val;
  } else if (idx < NQ + NK) {
    int j = idx - NQ;
    int u = j % UUP, row = j / UUP;
    float val = 0.f;
    if (u < UU) {
      const float* x = items + row*EE;
      for (int k = 0; k < EE; ++k) val += x[k] * Wk[k*UUP + u];
    }
    Kp[j] = val;
  }
}

// ---------------- trunk input packing: profile | context ----------------
__global__ __launch_bounds__(256) void packx_kernel(const float* __restrict__ up,
                                                    const float* __restrict__ cx,
                                                    __bf16* __restrict__ Xbf) {
  int idx = blockIdx.x * 256 + threadIdx.x;   // ROWS * 192
  if (idx >= ROWS * 192) return;
  int c = idx % 192, row = idx / 192;
  int b = row / TQ;
  float v = (c < 128) ? up[b*128 + c] : cx[b*64 + (c - 128)];
  Xbf[row*256 + 64 + c] = (__bf16)v;
}

// ---------------- attention + interest: one block per (b,tq) ----------------
__global__ __launch_bounds__(256) void attn_kernel(const float* __restrict__ ub,
                                                   const float* __restrict__ items,
                                                   const float* __restrict__ Qp,
                                                   const float* __restrict__ Kp,
                                                   const __bf16* __restrict__ WpF,
                                                   const float* __restrict__ W2p,
                                                   const float* __restrict__ da,
                                                   const float* __restrict__ dm,
                                                   const float* __restrict__ dv,
                                                   const float* __restrict__ b2,
                                                   __bf16* __restrict__ Xbf) {
  __shared__ __align__(32) float s_ub[TTP * EE];   // history tile, fp32
  __shared__ __align__(32) float s_k[EE];          // candidate item vector
  __shared__ float s_w[TTP];                       // attention weight accumulators

  int b = blockIdx.x / TQ, tq = blockIdx.x % TQ;
  int tid = threadIdx.x;

  for (int i = tid; i < TTP*EE; i += 256) {
    int t = i >> 6, e = i & 63;
    s_ub[i] = (t < TT) ? ub[(b*TT + t)*EE + e] : 0.f;
  }
  if (tid < EE) s_k[tid] = items[(b*TQ + tq)*EE + tid];
  for (int i = tid; i < TTP; i += 256) s_w[i] = b2[0];
  __syncthreads();

  int wave = tid >> 5, lane = tid & 31;
  int lhalf = lane >> 4, lmod = lane & 15;
  const int NTILES = (TTP/16) * (UUP/16);   // 13 * 3 = 39

  for (int tile = wave; tile < NTILES; tile += 8) {
    int mt = tile / 3, nt = tile % 3;
    int m0 = mt * 16, n0 = nt * 16;
    const float* ubrow = s_ub + (m0 + lmod)*EE;
    v8f acc = {};
    #pragma unroll
    for (int ks = 0; ks < 2; ++ks) {
      int kb = ks * 32;
      // A fragment: two contiguous K-runs of 8 (ISA 16-bit A layout), q .* k in fp32 -> bf16
      v8f q1 = *(const v8f*)(ubrow + kb + 8*lhalf);
      v8f q2 = *(const v8f*)(ubrow + kb + 16 + 8*lhalf);
      v8f k1 = *(const v8f*)(s_k + kb + 8*lhalf);
      v8f k2 = *(const v8f*)(s_k + kb + 16 + 8*lhalf);
      v8f p1 = q1 * k1, p2 = q2 * k2;
      v16bf afrag;
      #pragma unroll
      for (int j = 0; j < 8; ++j) {
        afrag[j]     = (__bf16)p1[j];
        afrag[8 + j] = (__bf16)p2[j];
      }
      // B fragment: pre-swizzled, one aligned 32B load per lane
      v16bf bfrag = *(const v16bf*)(WpF + ((nt*2 + ks)*32 + lane)*16);
      acc = __builtin_amdgcn_wmma_f32_16x16x32_bf16(
          false, afrag, false, bfrag, (short)0, acc, false, false);
    }
    // epilogue: h = P + Q' + K' (b1 folded into Q'); dice; * W2; reduce into s_w
    int n = n0 + lmod;
    float kpv   = Kp[(b*TQ + tq)*UUP + n];
    float rstd  = __builtin_amdgcn_rsqf(dv[n] + 1e-6f);
    float alpha = da[n], mean = dm[n], w2v = W2p[n];
    #pragma unroll
    for (int r = 0; r < 8; ++r) {
      int m = m0 + r + 8*lhalf;
      float h = acc[r] + Qp[(b*TTP + m)*UUP + n] + kpv;
      float z = (h - mean) * rstd;
      float p = __builtin_amdgcn_rcpf(1.f + __builtin_amdgcn_exp2f(-z * 1.44269504f));
      float val = (p + alpha * (1.f - p)) * h;
      atomicAdd(&s_w[m], val * w2v);
    }
  }
  __syncthreads();

  // interest[e] = sum_t w[t] * ub[t][e]  -> bf16 into trunk input row
  if (tid < EE) {
    float acc2 = 0.f;
    for (int t = 0; t < TT; ++t) acc2 += s_w[t] * s_ub[t*EE + tid];
    Xbf[(b*TQ + tq)*256 + tid] = (__bf16)acc2;
  }
}

// ---------------- trunk GEMM layer: relu(X@W' + b'), one wave per 16x16 tile ----------------
template<int K, int N>
__global__ __launch_bounds__(128) void trunk_kernel(const __bf16* __restrict__ X,
                                                    const __bf16* __restrict__ Wf,
                                                    const float* __restrict__ bias,
                                                    __bf16* __restrict__ Ybf,
                                                    float* __restrict__ Yf32) {
  constexpr int NT = N / 16;
  int waveGlobal = blockIdx.x * 4 + (threadIdx.x >> 5);
  int mt = waveGlobal / NT, nt = waveGlobal % NT;
  int m0 = mt * 16, n0 = nt * 16;
  int lane = threadIdx.x & 31, lhalf = lane >> 4, lmod = lane & 15;

  const __bf16* xrow = X + (m0 + lmod)*K;
  v8f acc = {};
  #pragma unroll
  for (int kt = 0; kt < K/32; ++kt) {
    int kb = kt * 32;
    v8bf a1 = *(const v8bf*)(xrow + kb + 8*lhalf);          // 16B aligned contiguous run
    v8bf a2 = *(const v8bf*)(xrow + kb + 16 + 8*lhalf);
    v16bf afrag = __builtin_shufflevector(a1, a2,
        0,1,2,3,4,5,6,7,8,9,10,11,12,13,14,15);
    v16bf bfrag = *(const v16bf*)(Wf + ((kt*NT + nt)*32 + lane)*16);  // 32B aligned
    acc = __builtin_amdgcn_wmma_f32_16x16x32_bf16(
        false, afrag, false, bfrag, (short)0, acc, false, false);
  }
  float bv = bias[n0 + lmod];
  #pragma unroll
  for (int r = 0; r < 8; ++r) {
    int m = m0 + r + 8*lhalf;
    float v = acc[r] + bv;
    v = v > 0.f ? v : 0.f;
    if (Ybf)  Ybf[m*N + n0 + lmod] = (__bf16)v;
    if (Yf32) Yf32[m*N + n0 + lmod] = v;
  }
}

// ---------------- host ----------------
extern "C" void kernel_launch(void* const* d_in, const int* in_sizes, int n_in,
                              void* d_out, int out_size, void* d_ws, size_t ws_size,
                              hipStream_t stream) {
  const float* ub     = (const float*)d_in[0];
  const float* items  = (const float*)d_in[1];
  const float* uprof  = (const float*)d_in[2];
  const float* ctx    = (const float*)d_in[3];
  const float* W1     = (const float*)d_in[4];
  const float* b1     = (const float*)d_in[5];
  const float* dice_a = (const float*)d_in[6];
  const float* dice_m = (const float*)d_in[7];
  const float* dice_v = (const float*)d_in[8];
  const float* W2     = (const float*)d_in[9];
  const float* b2     = (const float*)d_in[10];
  const float* bn1_g = (const float*)d_in[11], *bn1_b = (const float*)d_in[12];
  const float* bn1_m = (const float*)d_in[13], *bn1_v = (const float*)d_in[14];
  const float* Wm1   = (const float*)d_in[15], *bm1   = (const float*)d_in[16];
  const float* bn2_g = (const float*)d_in[17], *bn2_b = (const float*)d_in[18];
  const float* bn2_m = (const float*)d_in[19], *bn2_v = (const float*)d_in[20];
  const float* Wm2   = (const float*)d_in[21], *bm2   = (const float*)d_in[22];
  const float* bn3_g = (const float*)d_in[23], *bn3_b = (const float*)d_in[24];
  const float* bn3_m = (const float*)d_in[25], *bn3_v = (const float*)d_in[26];
  const float* Wm3   = (const float*)d_in[27], *bm3   = (const float*)d_in[28];

  char* p = (char*)d_ws;
  auto alloc = [&](size_t bytes) -> void* {
    void* r = (void*)p;
    p += (bytes + 255) & ~(size_t)255;
    return r;
  };
  float* Wq   = (float*)alloc(EE*UUP*4);
  float* Wk   = (float*)alloc(EE*UUP*4);
  float* W2p  = (float*)alloc(UUP*4);
  float* da   = (float*)alloc(UUP*4);
  float* dm   = (float*)alloc(UUP*4);
  float* dvv  = (float*)alloc(UUP*4);
  float* Qp   = (float*)alloc((size_t)BB*TTP*UUP*4);
  float* Kp   = (float*)alloc((size_t)BB*TQ*UUP*4);
  float* b1p  = (float*)alloc(256*4);
  float* b2p  = (float*)alloc(128*4);
  float* b3p  = (float*)alloc(64*4);
  __bf16* WpF  = (__bf16*)alloc(3*2*32*16*2);
  __bf16* W1p  = (__bf16*)alloc(256*256*2);
  __bf16* W2mp = (__bf16*)alloc(256*128*2);
  __bf16* W3p  = (__bf16*)alloc(128*64*2);
  __bf16* Xbf  = (__bf16*)alloc((size_t)ROWS*256*2);
  __bf16* Y1   = (__bf16*)alloc((size_t)ROWS*256*2);
  __bf16* Y2   = (__bf16*)alloc((size_t)ROWS*128*2);

  PrepArgs A;
  A.W1 = W1; A.W2 = W2; A.dice_a = dice_a; A.dice_m = dice_m; A.dice_v = dice_v;
  A.bn1_g = bn1_g; A.bn1_b = bn1_b; A.bn1_m = bn1_m; A.bn1_v = bn1_v; A.Wm1 = Wm1; A.bm1 = bm1;
  A.bn2_g = bn2_g; A.bn2_b = bn2_b; A.bn2_m = bn2_m; A.bn2_v = bn2_v; A.Wm2 = Wm2; A.bm2 = bm2;
  A.bn3_g = bn3_g; A.bn3_b = bn3_b; A.bn3_m = bn3_m; A.bn3_v = bn3_v; A.Wm3 = Wm3; A.bm3 = bm3;
  A.Wq = Wq; A.Wk = Wk; A.W2p = W2p; A.da = da; A.dm = dm; A.dv = dvv;
  A.b1p = b1p; A.b2p = b2p; A.b3p = b3p;
  A.WpF = WpF; A.W1p = W1p; A.W2mp = W2mp; A.W3p = W3p;

  prep_kernel<<<1, 256, 0, stream>>>(A);

  const int qk_total = BB*TTP*UUP + BB*TQ*UUP;          // 396288
  qk_kernel<<<(qk_total + 255)/256, 256, 0, stream>>>(ub, items, Wq, Wk, b1, Qp, Kp);

  packx_kernel<<<(ROWS*192 + 255)/256, 256, 0, stream>>>(uprof, ctx, Xbf);

  attn_kernel<<<BB*TQ, 256, 0, stream>>>(ub, items, Qp, Kp, WpF, W2p, da, dm, dvv, b2, Xbf);

  // trunk: relu chains with BN folded into fragment-swizzled weights
  trunk_kernel<256,256><<<(ROWS/16)*(256/16)/4, 128, 0, stream>>>(Xbf, W1p,  b1p, Y1, nullptr);
  trunk_kernel<256,128><<<(ROWS/16)*(128/16)/4, 128, 0, stream>>>(Y1,  W2mp, b2p, Y2, nullptr);
  trunk_kernel<128, 64><<<(ROWS/16)*( 64/16)/4, 128, 0, stream>>>(Y2,  W3p,  b3p, nullptr, (float*)d_out);
}
